// KGLinkPredictor_80144089743681
// MI455X (gfx1250) — matrix operand order repaired and checked by reading, not compiled, not run on hardware
//
#include <hip/hip_runtime.h>
#include <hip/hip_bf16.h>

typedef __attribute__((ext_vector_type(16))) _Float16 v16h;
typedef __attribute__((ext_vector_type(8)))  float    v8f;

#define N_NODES 50000
#define N_EDGES 600000
#define HID     128
#define N_TILES (N_NODES / 16)   // 3125 exactly

// ---------------- helpers ----------------

__device__ __forceinline__ void atomAddF(float* p, float v) {
    // lowers to GLOBAL_ATOMIC_ADD_F32 (no return -> STOREcnt path)
    unsafeAtomicAdd(p, v);
}

// ---------------- preprocessing kernels ----------------

__global__ void deg_init_kernel(float* deg, int n) {
    int i = blockIdx.x * blockDim.x + threadIdx.x;
    if (i < n) deg[i] = 1.0f;             // self-loop contributes 1
}

__global__ void deg_count_kernel(const int* __restrict__ dst, float* deg, int e) {
    int i = blockIdx.x * blockDim.x + threadIdx.x;
    if (i < e) atomAddF(&deg[dst[i]], 1.0f);
}

__global__ void deg_rsqrt_kernel(float* deg, int n) {
    int i = blockIdx.x * blockDim.x + threadIdx.x;
    if (i < n) deg[i] = rsqrtf(deg[i]);   // in-place: deg -> dinv
}

__global__ void edge_norm_kernel(const int* __restrict__ src, const int* __restrict__ dst,
                                 const float* __restrict__ dinv, float* __restrict__ norm, int e) {
    int i = blockIdx.x * blockDim.x + threadIdx.x;
    if (i < e) norm[i] = dinv[src[i]] * dinv[dst[i]];
}

__global__ void w_to_half_kernel(const float* __restrict__ w, _Float16* __restrict__ wh, int n) {
    int i = blockIdx.x * blockDim.x + threadIdx.x;
    if (i < n) wh[i] = (_Float16)w[i];
}

// ---------------- WMMA GEMM: H[N,128] = act(X[N,128]) @ W[128,128] ----------------
// Block = 256 threads = 8 waves. Wave w owns output columns [16w, 16w+16).
// B fragments (4 k-blocks of 32) hoisted into registers once per wave,
// then grid-stride over 16-row tiles: 4 x v_wmma_f32_16x16x32_f16 per tile.

template <bool RELU>
__global__ void gemm_wmma_kernel(const float* __restrict__ X,
                                 const _Float16* __restrict__ Wh,
                                 float* __restrict__ H, int nTiles) {
    const int wave  = threadIdx.x >> 5;
    const int lane  = threadIdx.x & 31;
    const int mrow  = lane & 15;          // row (A) / col (B,D) index within tile
    const int khalf = lane >> 4;          // lane-half selects K sub-range
    const int col0  = wave * 16;

    // --- load B fragments: B is 32x16 per k-block; col = lane%16,
    //     K = kb*32 + khalf*16 + elem  (elem 0..15 in order) ---
    v16h bfrag[4];
#pragma unroll
    for (int kb = 0; kb < 4; ++kb) {
        const _Float16* wp = Wh + (size_t)(kb * 32 + khalf * 16) * HID + col0 + mrow;
#pragma unroll
        for (int e = 0; e < 16; ++e) bfrag[kb][e] = wp[(size_t)e * HID];
    }

    for (int rt = blockIdx.x; rt < nTiles; rt += gridDim.x) {
        const float* xp = X + (size_t)(rt * 16 + mrow) * HID;
        v8f acc = {};
#pragma unroll
        for (int kb = 0; kb < 4; ++kb) {
            // A is 16x32: row = lane%16; elems 0..15 -> K = base+0..15,
            // base = kb*32 + khalf*16
            const int base = kb * 32 + khalf * 16;
            v16h a;
#pragma unroll
            for (int e = 0; e < 8; ++e) {
                float f0 = xp[base + e];
                float f1 = xp[base + 8 + e];
                if (RELU) { f0 = fmaxf(f0, 0.0f); f1 = fmaxf(f1, 0.0f); }
                a[e]     = (_Float16)f0;
                a[e + 8] = (_Float16)f1;
            }
            acc = __builtin_amdgcn_wmma_f32_16x16x32_f16(
                      false, a, false, bfrag[kb], (short)0, acc, false, false);
        }
        // D: VGPR r holds M = khalf*8 + r, N = col0 + lane%16
        float* hp = H + (size_t)(rt * 16 + khalf * 8) * HID + col0 + mrow;
#pragma unroll
        for (int r = 0; r < 8; ++r) hp[(size_t)r * HID] = acc[r];
    }
}

// ---------------- out[i] = bias + h[i] * dinv[i]^2  (self-loop folded in) ----------------
// one thread per (node, float4-chunk): N*32 threads

__global__ void init_out_kernel(const float* __restrict__ H, const float* __restrict__ dinv,
                                const float* __restrict__ bias, float* __restrict__ Xout, int n) {
    int gid  = blockIdx.x * blockDim.x + threadIdx.x;
    int i    = gid >> 5;
    int lane = gid & 31;
    if (i >= n) return;
    float di = dinv[i];
    float s  = di * di;
    int f = lane * 4;
    const float4 h = *(const float4*)(H + (size_t)i * HID + f);
    const float4 b = *(const float4*)(bias + f);
    float4 o;
    o.x = b.x + h.x * s; o.y = b.y + h.y * s;
    o.z = b.z + h.z * s; o.w = b.w + h.w * s;
    *(float4*)(Xout + (size_t)i * HID + f) = o;
}

// ---------------- edge scatter: Xout[dst] += H[src] * norm ----------------
// one wave per edge; lane handles float4 at feature lane*4; atomics resolve in L2

__global__ void scatter_kernel(const float* __restrict__ H,
                               const int* __restrict__ src, const int* __restrict__ dst,
                               const float* __restrict__ norm,
                               float* __restrict__ Xout, int e) {
    int gid  = blockIdx.x * blockDim.x + threadIdx.x;
    int edge = gid >> 5;
    int lane = gid & 31;
    if (edge >= e) return;
    int   s  = src[edge];
    int   d  = dst[edge];
    float nv = norm[edge];
    const float4 h = *(const float4*)(H + (size_t)s * HID + lane * 4);
    float* op = Xout + (size_t)d * HID + lane * 4;
    atomAddF(op + 0, h.x * nv);
    atomAddF(op + 1, h.y * nv);
    atomAddF(op + 2, h.z * nv);
    atomAddF(op + 3, h.w * nv);
}

// ---------------- host side ----------------

static inline size_t alignup(size_t x, size_t a) { return (x + a - 1) & ~(a - 1); }

extern "C" void kernel_launch(void* const* d_in, const int* in_sizes, int n_in,
                              void* d_out, int out_size, void* d_ws, size_t ws_size,
                              hipStream_t stream) {
    const int N = N_NODES, E = N_EDGES;

    const int*   edge = (const int*)d_in[0];
    const int*   src  = edge;
    const int*   dst  = edge + E;
    const float* X0   = (const float*)d_in[1];
    const float* W[3] = { (const float*)d_in[2], (const float*)d_in[4], (const float*)d_in[6] };
    const float* B[3] = { (const float*)d_in[3], (const float*)d_in[5], (const float*)d_in[7] };

    // workspace carve-up (~52 MB total)
    char* ws = (char*)d_ws;
    size_t off = 0;
    float* dinv = (float*)(ws + off);  off = alignup(off + (size_t)N * 4, 256);
    float* norm = (float*)(ws + off);  off = alignup(off + (size_t)E * 4, 256);
    _Float16* Wh = (_Float16*)(ws + off); off = alignup(off + (size_t)3 * HID * HID * 2, 256);
    float* bufH = (float*)(ws + off);  off = alignup(off + (size_t)N * HID * 4, 256);
    float* buf0 = (float*)(ws + off);  off = alignup(off + (size_t)N * HID * 4, 256);
    (void)ws_size;

    const int T = 256;

    // ---- graph preprocessing (cheap, done every call: deterministic) ----
    deg_init_kernel <<<(N + T - 1) / T, T, 0, stream>>>(dinv, N);
    deg_count_kernel<<<(E + T - 1) / T, T, 0, stream>>>(dst, dinv, E);
    deg_rsqrt_kernel<<<(N + T - 1) / T, T, 0, stream>>>(dinv, N);
    edge_norm_kernel<<<(E + T - 1) / T, T, 0, stream>>>(src, dst, dinv, norm, E);
    for (int l = 0; l < 3; ++l)
        w_to_half_kernel<<<(HID * HID + T - 1) / T, T, 0, stream>>>(W[l], Wh + (size_t)l * HID * HID, HID * HID);

    const int gemmBlocks    = 625;                 // 5 row-tiles per block (grid-stride)
    const int initBlocks    = (N * 32 + T - 1) / T; // 6250
    const int scatterBlocks = (E * 32 + T - 1) / T; // 75000

    // ---- layer 0: X0 -> bufH -> buf0 ----
    gemm_wmma_kernel<false><<<gemmBlocks, T, 0, stream>>>(X0, Wh, bufH, N_TILES);
    init_out_kernel<<<initBlocks, T, 0, stream>>>(bufH, dinv, B[0], buf0, N);
    scatter_kernel<<<scatterBlocks, T, 0, stream>>>(bufH, src, dst, norm, buf0, E);

    // ---- layer 1: relu(buf0) -> bufH -> buf0 ----
    gemm_wmma_kernel<true><<<gemmBlocks, T, 0, stream>>>(buf0, Wh + (size_t)1 * HID * HID, bufH, N_TILES);
    init_out_kernel<<<initBlocks, T, 0, stream>>>(bufH, dinv, B[1], buf0, N);
    scatter_kernel<<<scatterBlocks, T, 0, stream>>>(bufH, src, dst, norm, buf0, E);

    // ---- layer 2: relu(buf0) -> bufH -> d_out ----
    float* out = (float*)d_out;
    gemm_wmma_kernel<true><<<gemmBlocks, T, 0, stream>>>(buf0, Wh + (size_t)2 * HID * HID, bufH, N_TILES);
    init_out_kernel<<<initBlocks, T, 0, stream>>>(bufH, dinv, B[2], out, N);
    scatter_kernel<<<scatterBlocks, T, 0, stream>>>(bufH, src, dst, norm, out, E);
}
